// Self_Attention_19172734009610
// MI455X (gfx1250) — compile-verified, matrix operand order checked by reading, not run on hardware
//
#include <hip/hip_runtime.h>
#include <hip/hip_bf16.h>

typedef __attribute__((ext_vector_type(16))) _Float16 v16h;
typedef __attribute__((ext_vector_type(8)))  _Float16 v8h;
typedef __attribute__((ext_vector_type(8)))  float    v8f;

#define Bb 4
#define CC 256
#define DD 32
#define NN 4096

// Load one f16 A/B operand for V_WMMA_F32_16X16X32_F16 from K-contiguous
// storage: lane<16 needs K in {0..7, 16..23}, lane>=16 needs {8..15, 24..31}
// -> two 16B vector loads per lane.
__device__ __forceinline__ v16h load_ab(const _Float16* rowbase, int hi8) {
    union { v16h v; v8h p[2]; } u;
    u.p[0] = *(const v8h*)(rowbase + hi8);
    u.p[1] = *(const v8h*)(rowbase + 16 + hi8);
    return u.v;
}

__device__ __forceinline__ v8f wmma16(v16h a, v16h b, v8f c) {
    return __builtin_amdgcn_wmma_f32_16x16x32_f16(false, a, false, b,
                                                  (short)0, c, false, false);
}

// ---------------------------------------------------------------------------
// Kernel 1: convert weights to f16 (Wf/Wg/Wh: [32][256], Wv: [256][32];
// all already K-contiguous for their WMMA role).
// ---------------------------------------------------------------------------
__global__ void __launch_bounds__(256)
wconv_kernel(const float* __restrict__ Wf, const float* __restrict__ Wg,
             const float* __restrict__ Wh, const float* __restrict__ Wv,
             _Float16* __restrict__ Wfh, _Float16* __restrict__ Wgh,
             _Float16* __restrict__ Whh, _Float16* __restrict__ Wvh) {
    int t = blockIdx.x * 256 + threadIdx.x;   // 0 .. 32767
    int a = t >> 13;                          // which matrix
    int i = t & 8191;
    if (a == 0)      Wfh[i] = (_Float16)Wf[i];
    else if (a == 1) Wgh[i] = (_Float16)Wg[i];
    else if (a == 2) Whh[i] = (_Float16)Wh[i];
    else             Wvh[i] = (_Float16)Wv[i];
}

// ---------------------------------------------------------------------------
// Kernel 2: x [b][c][n] f32  ->  Xt [b][n][c] f16  (LDS 32x32 tile transpose)
// ---------------------------------------------------------------------------
__global__ void __launch_bounds__(256)
xtrans_kernel(const float* __restrict__ x, _Float16* __restrict__ Xt) {
    __shared__ float tb[32][33];
    int n0 = blockIdx.x * 32, c0 = blockIdx.y * 32, b = blockIdx.z;
    int tx = threadIdx.x, ty = threadIdx.y;           // (32, 8)
    #pragma unroll
    for (int i = 0; i < 4; ++i)
        tb[ty + i * 8][tx] = x[((size_t)b * CC + c0 + ty + i * 8) * NN + n0 + tx];
    __syncthreads();
    #pragma unroll
    for (int i = 0; i < 4; ++i)
        Xt[((size_t)b * NN + n0 + ty + i * 8) * CC + c0 + tx] =
            (_Float16)tb[tx][ty + i * 8];
}

// ---------------------------------------------------------------------------
// Kernel 3: WMMA projections.  p = blockIdx.y selects f/g/h.
// One wave: 32 out-rows x 32 tokens, K-loop over C=256 (8 iters x 4 WMMAs).
// f,g stored token-major [n][32]; h stored d-major [d][n] (sa A-operand).
// ---------------------------------------------------------------------------
__global__ void __launch_bounds__(128)
proj_wmma_kernel(const _Float16* __restrict__ Xt,
                 const _Float16* __restrict__ Wfh, const _Float16* __restrict__ Wgh,
                 const _Float16* __restrict__ Whh,
                 const float* __restrict__ bf, const float* __restrict__ bg,
                 const float* __restrict__ bh,
                 _Float16* __restrict__ Ft, _Float16* __restrict__ Gt,
                 _Float16* __restrict__ Hd) {
    int lane = threadIdx.x & 31, wave = threadIdx.x >> 5;
    int p = blockIdx.y, b = blockIdx.z;
    int m0 = (blockIdx.x * 4 + wave) * 32;
    int col = lane & 15, hi = (lane >= 16) ? 8 : 0;
    const _Float16* W    = (p == 0) ? Wfh : ((p == 1) ? Wgh : Whh);
    const float*    bias = (p == 0) ? bf  : ((p == 1) ? bg  : bh);
    const _Float16* Xb = Xt + (size_t)b * NN * CC;

    v8f acc[2][2] = {};
    for (int k0 = 0; k0 < CC; k0 += 32) {
        v16h a0 = load_ab(W + (size_t)col * CC + k0, hi);
        v16h a1 = load_ab(W + (size_t)(16 + col) * CC + k0, hi);
        v16h b0 = load_ab(Xb + (size_t)(m0 + col) * CC + k0, hi);
        v16h b1 = load_ab(Xb + (size_t)(m0 + 16 + col) * CC + k0, hi);
        acc[0][0] = wmma16(a0, b0, acc[0][0]);
        acc[0][1] = wmma16(a0, b1, acc[0][1]);
        acc[1][0] = wmma16(a1, b0, acc[1][0]);
        acc[1][1] = wmma16(a1, b1, acc[1][1]);
    }
    float bia[2][8];
    #pragma unroll
    for (int dt = 0; dt < 2; ++dt) {
        float4 bA = *(const float4*)(bias + dt * 16 + hi);
        float4 bB = *(const float4*)(bias + dt * 16 + hi + 4);
        bia[dt][0] = bA.x; bia[dt][1] = bA.y; bia[dt][2] = bA.z; bia[dt][3] = bA.w;
        bia[dt][4] = bB.x; bia[dt][5] = bB.y; bia[dt][6] = bB.z; bia[dt][7] = bB.w;
    }
    if (p < 2) {
        _Float16* dst = ((p == 0) ? Ft : Gt) + (size_t)b * NN * DD;
        #pragma unroll
        for (int dt = 0; dt < 2; ++dt)
            #pragma unroll
            for (int mt = 0; mt < 2; ++mt) {
                v8h pk;
                #pragma unroll
                for (int r = 0; r < 8; ++r)
                    pk[r] = (_Float16)(acc[dt][mt][r] + bia[dt][r]);
                *(v8h*)(dst + (size_t)(m0 + mt * 16 + col) * DD + dt * 16 + hi) = pk;
            }
    } else {
        _Float16* dst = Hd + (size_t)b * DD * NN;
        #pragma unroll
        for (int dt = 0; dt < 2; ++dt)
            #pragma unroll
            for (int mt = 0; mt < 2; ++mt)
                #pragma unroll
                for (int r = 0; r < 8; ++r)
                    dst[(size_t)(dt * 16 + hi + r) * NN + m0 + mt * 16 + col] =
                        (_Float16)(acc[dt][mt][r] + bia[dt][r]);
    }
}

// ---------------------------------------------------------------------------
// Kernel 4: softmax stats, two-pass.  Wave owns 32 rows n, sweeps all m tiles.
// Pass 1: logits -> running max (8 v_max / WMMA).  Cross-lane max merge.
// Pass 2: logits -> sum exp(l - max).  Cross-lane sum.  S[b][n] = {max, 1/Z}.
// ---------------------------------------------------------------------------
__global__ void __launch_bounds__(128)
stats_kernel(const _Float16* __restrict__ Ft, const _Float16* __restrict__ Gt,
             float* __restrict__ S) {
    int lane = threadIdx.x & 31, wave = threadIdx.x >> 5;
    int b = blockIdx.y;
    int n0 = (blockIdx.x * 4 + wave) * 32;
    int col = lane & 15, hi = (lane >= 16) ? 8 : 0;
    const _Float16* Fb = Ft + (size_t)b * NN * DD;
    const _Float16* Gb = Gt + (size_t)b * NN * DD;

    v16h a0 = load_ab(Fb + (size_t)(n0 + col) * DD, hi);
    v16h a1 = load_ab(Fb + (size_t)(n0 + 16 + col) * DD, hi);

    float mx[2][8];
    #pragma unroll
    for (int t = 0; t < 2; ++t)
        #pragma unroll
        for (int r = 0; r < 8; ++r) mx[t][r] = -3.0e38f;

    for (int m0 = 0; m0 < NN; m0 += 16) {
        v16h bb = load_ab(Gb + (size_t)(m0 + col) * DD, hi);
        v8f z = {};
        v8f c0 = wmma16(a0, bb, z);
        v8f c1 = wmma16(a1, bb, z);
        #pragma unroll
        for (int r = 0; r < 8; ++r) {
            mx[0][r] = fmaxf(mx[0][r], c0[r]);
            mx[1][r] = fmaxf(mx[1][r], c1[r]);
        }
    }
    #pragma unroll
    for (int t = 0; t < 2; ++t)
        #pragma unroll
        for (int r = 0; r < 8; ++r)
            #pragma unroll
            for (int off = 1; off < 16; off <<= 1)
                mx[t][r] = fmaxf(mx[t][r], __shfl_xor(mx[t][r], off, 16));

    float sm[2][8] = {};
    for (int m0 = 0; m0 < NN; m0 += 16) {
        v16h bb = load_ab(Gb + (size_t)(m0 + col) * DD, hi);
        v8f z = {};
        v8f c0 = wmma16(a0, bb, z);
        v8f c1 = wmma16(a1, bb, z);
        #pragma unroll
        for (int r = 0; r < 8; ++r) {
            sm[0][r] += __expf(c0[r] - mx[0][r]);
            sm[1][r] += __expf(c1[r] - mx[1][r]);
        }
    }
    #pragma unroll
    for (int t = 0; t < 2; ++t)
        #pragma unroll
        for (int r = 0; r < 8; ++r)
            #pragma unroll
            for (int off = 1; off < 16; off <<= 1)
                sm[t][r] += __shfl_xor(sm[t][r], off, 16);

    if (col == 0) {
        #pragma unroll
        for (int t = 0; t < 2; ++t)
            #pragma unroll
            for (int r = 0; r < 8; ++r) {
                int row = n0 + t * 16 + hi + r;
                S[((size_t)b * NN + row) * 2]     = mx[t][r];
                S[((size_t)b * NN + row) * 2 + 1] = 1.0f / sm[t][r];
            }
    }
}

// ---------------------------------------------------------------------------
// Kernel 5: attention apply.  Wave owns 32 output tokens m, sa[32][32] in four
// accumulators.  Per 32-n chunk: 4 logit WMMAs -> exp -> packed ds_store_b128
// into m-major LDS P tile -> 4 sa WMMAs from ds_load_b128 operands.
// Result stored token-major f16 SAt[b][m][32].
// ---------------------------------------------------------------------------
__global__ void __launch_bounds__(128)
attn_out_kernel(const _Float16* __restrict__ Ft, const _Float16* __restrict__ Gt,
                const _Float16* __restrict__ Hd, const float* __restrict__ S,
                _Float16* __restrict__ SAt) {
    __shared__ __align__(16) _Float16 Pt[4][2][16][40];  // [wave][mt][m][n], padded
    int lane = threadIdx.x & 31, wave = threadIdx.x >> 5;
    int b = blockIdx.y;
    int m0 = (blockIdx.x * 4 + wave) * 32;
    int col = lane & 15, hi = (lane >= 16) ? 8 : 0;
    const _Float16* Fb = Ft + (size_t)b * NN * DD;
    const _Float16* Gb = Gt + (size_t)b * NN * DD;
    const _Float16* Hb = Hd + (size_t)b * DD * NN;
    const float*    Sb = S + (size_t)b * NN * 2;

    v16h gB0 = load_ab(Gb + (size_t)(m0 + col) * DD, hi);
    v16h gB1 = load_ab(Gb + (size_t)(m0 + 16 + col) * DD, hi);

    v8f acc[2][2] = {};   // [d-tile][m-tile]
    for (int n0 = 0; n0 < NN; n0 += 32) {
        #pragma unroll
        for (int t = 0; t < 2; ++t) {
            v16h fA = load_ab(Fb + (size_t)(n0 + t * 16 + col) * DD, hi);
            const float* sp = Sb + (size_t)(n0 + t * 16 + hi) * 2;
            float4 s0 = *(const float4*)(sp);
            float4 s1 = *(const float4*)(sp + 4);
            float4 s2 = *(const float4*)(sp + 8);
            float4 s3 = *(const float4*)(sp + 12);
            float rm[8] = {s0.x, s0.z, s1.x, s1.z, s2.x, s2.z, s3.x, s3.z};
            float ri[8] = {s0.y, s0.w, s1.y, s1.w, s2.y, s2.w, s3.y, s3.w};
            #pragma unroll
            for (int mt = 0; mt < 2; ++mt) {
                v8f z = {};
                v8f c = wmma16(fA, mt ? gB1 : gB0, z);
                v8h pk;
                #pragma unroll
                for (int r = 0; r < 8; ++r)
                    pk[r] = (_Float16)(__expf(c[r] - rm[r]) * ri[r]);
                *(v8h*)&Pt[wave][mt][col][t * 16 + hi] = pk;
            }
        }
        __syncthreads();
        v16h hA0 = load_ab(Hb + (size_t)col * NN + n0, hi);
        v16h hA1 = load_ab(Hb + (size_t)(16 + col) * NN + n0, hi);
        v16h pB0 = load_ab(&Pt[wave][0][col][0], hi);
        v16h pB1 = load_ab(&Pt[wave][1][col][0], hi);
        acc[0][0] = wmma16(hA0, pB0, acc[0][0]);
        acc[0][1] = wmma16(hA0, pB1, acc[0][1]);
        acc[1][0] = wmma16(hA1, pB0, acc[1][0]);
        acc[1][1] = wmma16(hA1, pB1, acc[1][1]);
        __syncthreads();
    }
    _Float16* dst = SAt + (size_t)b * NN * DD;
    #pragma unroll
    for (int dt = 0; dt < 2; ++dt)
        #pragma unroll
        for (int mt = 0; mt < 2; ++mt) {
            v8h pk;
            #pragma unroll
            for (int r = 0; r < 8; ++r) pk[r] = (_Float16)acc[dt][mt][r];
            *(v8h*)(dst + (size_t)(m0 + mt * 16 + col) * DD + dt * 16 + hi) = pk;
        }
}

// ---------------------------------------------------------------------------
// Kernel 6: output 1x1 conv D->C via WMMA (K=32 -> single WMMA per tile),
// + bias, * gamma.  Wave: 32 c-rows x 32 tokens = 4 WMMAs.
// ---------------------------------------------------------------------------
__global__ void __launch_bounds__(128)
outw_kernel(const _Float16* __restrict__ Wvh, const _Float16* __restrict__ SAt,
            const float* __restrict__ bv, const float* __restrict__ gamma,
            float* __restrict__ out) {
    int lane = threadIdx.x & 31, wave = threadIdx.x >> 5;
    int b = blockIdx.z, c0 = blockIdx.y * 32;
    int m0 = (blockIdx.x * 4 + wave) * 32;
    int col = lane & 15, hi = (lane >= 16) ? 8 : 0;
    const _Float16* SAb = SAt + (size_t)b * NN * DD;

    v16h a0 = load_ab(Wvh + (size_t)(c0 + col) * DD, hi);
    v16h a1 = load_ab(Wvh + (size_t)(c0 + 16 + col) * DD, hi);
    v16h b0 = load_ab(SAb + (size_t)(m0 + col) * DD, hi);
    v16h b1 = load_ab(SAb + (size_t)(m0 + 16 + col) * DD, hi);
    v8f z = {};
    v8f cc[2][2];
    cc[0][0] = wmma16(a0, b0, z);
    cc[0][1] = wmma16(a0, b1, z);
    cc[1][0] = wmma16(a1, b0, z);
    cc[1][1] = wmma16(a1, b1, z);

    float gm = gamma[0];
    #pragma unroll
    for (int ct = 0; ct < 2; ++ct) {
        float4 bA = *(const float4*)(bv + c0 + ct * 16 + hi);
        float4 bB = *(const float4*)(bv + c0 + ct * 16 + hi + 4);
        float bia[8] = {bA.x, bA.y, bA.z, bA.w, bB.x, bB.y, bB.z, bB.w};
        #pragma unroll
        for (int mt = 0; mt < 2; ++mt)
            #pragma unroll
            for (int r = 0; r < 8; ++r) {
                int c = c0 + ct * 16 + hi + r;
                int m = m0 + mt * 16 + col;
                out[((size_t)b * CC + c) * NN + m] = gm * (cc[ct][mt][r] + bia[r]);
            }
    }
}

extern "C" void kernel_launch(void* const* d_in, const int* in_sizes, int n_in,
                              void* d_out, int out_size, void* d_ws, size_t ws_size,
                              hipStream_t stream) {
    const float* x     = (const float*)d_in[0];
    const float* Wf    = (const float*)d_in[1];
    const float* bf    = (const float*)d_in[2];
    const float* Wg    = (const float*)d_in[3];
    const float* bg    = (const float*)d_in[4];
    const float* Wh    = (const float*)d_in[5];
    const float* bh    = (const float*)d_in[6];
    const float* Wv    = (const float*)d_in[7];
    const float* bv    = (const float*)d_in[8];
    const float* gamma = (const float*)d_in[9];
    float* out = (float*)d_out;

    // workspace layout (~13 MB)
    char* w = (char*)d_ws;
    _Float16* Xt  = (_Float16*)w; w += (size_t)Bb * NN * CC * 2;   // 8 MB
    _Float16* Wfh = (_Float16*)w; w += 32 * CC * 2;
    _Float16* Wgh = (_Float16*)w; w += 32 * CC * 2;
    _Float16* Whh = (_Float16*)w; w += 32 * CC * 2;
    _Float16* Wvh = (_Float16*)w; w += CC * DD * 2;
    _Float16* Ft  = (_Float16*)w; w += (size_t)Bb * NN * DD * 2;   // 1 MB
    _Float16* Gt  = (_Float16*)w; w += (size_t)Bb * NN * DD * 2;
    _Float16* Hd  = (_Float16*)w; w += (size_t)Bb * DD * NN * 2;
    float*    S   = (float*)w;    w += (size_t)Bb * NN * 2 * 4;    // {max, 1/Z}
    _Float16* SAt = (_Float16*)w;

    wconv_kernel<<<128, 256, 0, stream>>>(Wf, Wg, Wh, Wv, Wfh, Wgh, Whh, Wvh);
    xtrans_kernel<<<dim3(NN / 32, CC / 32, Bb), dim3(32, 8), 0, stream>>>(x, Xt);
    proj_wmma_kernel<<<dim3(32, 3, Bb), 128, 0, stream>>>(
        Xt, Wfh, Wgh, Whh, bf, bg, bh, Ft, Gt, Hd);
    stats_kernel<<<dim3(32, Bb), 128, 0, stream>>>(Ft, Gt, S);
    attn_out_kernel<<<dim3(32, Bb), 128, 0, stream>>>(Ft, Gt, Hd, S, SAt);
    outw_kernel<<<dim3(32, 8, Bb), 128, 0, stream>>>(Wvh, SAt, bv, gamma, out);
}
